// TransEncoderLayer_59717225284215
// MI455X (gfx1250) — compile-verified, matrix operand order checked by reading, not run on hardware
//
#include <hip/hip_runtime.h>
#include <hip/hip_bf16.h>

#define BB 4
#define LL 4096
#define SSEQ 4096
#define DD 512
#define HH 8
#define DHD 64
#define NT (BB * LL)     /* 16384 token rows */
#define D2 (2 * DD)      /* 1024 */

typedef __bf16 v16bf __attribute__((ext_vector_type(16)));
typedef __bf16 v8bf  __attribute__((ext_vector_type(8)));
typedef float  v8f   __attribute__((ext_vector_type(8)));
typedef int    v4i   __attribute__((ext_vector_type(4)));

// ---------------------------------------------------------------------------
// CDNA5 async global->LDS copy (per-lane 16B DMA, tracked with ASYNCcnt)
// ---------------------------------------------------------------------------
#if defined(__has_builtin)
#if __has_builtin(__builtin_amdgcn_global_load_async_to_lds_b128)
#define HAVE_ASYNC_BUILTIN 1
#endif
#if __has_builtin(__builtin_amdgcn_s_wait_asynccnt)
#define HAVE_ASYNCWAIT_BUILTIN 1
#endif
#endif

static __device__ __forceinline__ void async_copy16(void* lds, const void* g) {
#ifdef HAVE_ASYNC_BUILTIN
  __builtin_amdgcn_global_load_async_to_lds_b128(
      (__attribute__((address_space(1))) v4i*)(g),
      (__attribute__((address_space(3))) v4i*)(lds), 0, 0);
#else
  asm volatile("global_load_async_to_lds_b128 %0, %1, off" ::"v"(
                   (unsigned)(size_t)lds),
               "v"(g)
               : "memory");
#endif
}

static __device__ __forceinline__ void wait_async0() {
#ifdef HAVE_ASYNCWAIT_BUILTIN
  __builtin_amdgcn_s_wait_asynccnt(0);
#else
  asm volatile("s_wait_asynccnt 0x0" ::: "memory");
#endif
}

// ---------------------------------------------------------------------------
// WMMA helpers
// ---------------------------------------------------------------------------
static __device__ __forceinline__ v8f wmma_bf16f32(v16bf a, v16bf b, v8f c) {
  // D = A(16x32) * B(32x16) + C, f32 accumulate
  return __builtin_amdgcn_wmma_f32_16x16x32_bf16(false, a, false, b, (short)0, c,
                                                 false, false);
}

// Load one 16x32 bf16 fragment from an LDS tile stored row-major with `stride`
// elements per row.  Per ISA 7.12.2 (16-bit A 16x32): lane&15 = row,
// (lane>>4)*8 = koff; vector elements 0..7 hold K=koff..koff+7, elements
// 8..15 hold K=16+koff..23+koff.  Two contiguous 16B LDS loads per lane.
static __device__ __forceinline__ v16bf frag_lds(const __bf16* tile, int stride,
                                                 int lane) {
  const int row  = lane & 15;
  const int koff = (lane >> 4) << 3;
  const __bf16* p = tile + row * stride + koff;
  v8bf lo = *(const v8bf*)(p);
  v8bf hi = *(const v8bf*)(p + 16);
  return __builtin_shufflevector(lo, hi, 0, 1, 2, 3, 4, 5, 6, 7, 8, 9, 10, 11,
                                 12, 13, 14, 15);
}

// ---------------------------------------------------------------------------
// Elementwise conversions
// ---------------------------------------------------------------------------
__global__ __launch_bounds__(256) void f32_to_bf16_kernel(
    const float* __restrict__ in, __bf16* __restrict__ out, size_t n, int relu) {
  size_t i = (size_t)blockIdx.x * 256 + threadIdx.x;
  if (i < n) {
    float v = in[i];
    if (relu) v = v > 0.f ? v : 0.f;
    out[i] = (__bf16)v;
  }
}

// W[k][n] (f32, KxN row-major) -> Wt[n][k] (bf16, NxK row-major)
__global__ __launch_bounds__(256) void transpose_bf16_kernel(
    const float* __restrict__ W, __bf16* __restrict__ Wt, int K, int N) {
  size_t i = (size_t)blockIdx.x * 256 + threadIdx.x;
  if (i < (size_t)K * N) {
    int k = (int)(i / N);
    int n = (int)(i % N);
    Wt[(size_t)n * K + k] = (__bf16)W[i];
  }
}

// ---------------------------------------------------------------------------
// Generic bf16 GEMM: C[M,N](f32) (+)= A[M,K](bf16,row-major) * Bt[N,K]^T
// Block = 256 threads = 8 waves; block tile 128x256, K-step 32, double-buffered
// LDS filled by async global->LDS DMA.  Wave (wr in 0..1, wc in 0..3) owns a
// 64x64 sub-tile = 4x4 wmma tiles: 16 WMMAs per 16 ds_load_b128 per K-step.
// If Cbf != nullptr the result is written bf16 to Cbf instead of f32 to C.
// ---------------------------------------------------------------------------
#define LDSTR 40 /* padded LDS row stride in bf16 elements (80B) */

__global__ __launch_bounds__(256) void gemm_bf16_kernel(
    const __bf16* __restrict__ A, int lda, const __bf16* __restrict__ Bt,
    int ldb, float* __restrict__ C, int ldc, int K, int accumulate,
    __bf16* __restrict__ Cbf) {
  __shared__ alignas(16) __bf16 As[2][128 * LDSTR];
  __shared__ alignas(16) __bf16 Bs[2][256 * LDSTR];

  const int tid  = threadIdx.x;
  const int lane = tid & 31;
  const int wid  = tid >> 5;
  const int wr   = wid >> 2;  // 0..1 (m)
  const int wc   = wid & 3;   // 0..3 (n)
  const size_t m0 = (size_t)blockIdx.y * 128;
  const size_t n0 = (size_t)blockIdx.x * 256;

  const __bf16* Ab = A + m0 * lda;
  const __bf16* Bb = Bt + n0 * ldb;

  auto stage_tiles = [&](int buf, int kk) {
#pragma unroll
    for (int i = 0; i < 2; ++i) {  // A tile: 128 rows x 32 cols
      int c   = tid + i * 256;     // 0..511 chunks of 8 bf16
      int row = c >> 2;            // 0..127
      int cc  = (c & 3) << 3;      // 0,8,16,24
      async_copy16(&As[buf][row * LDSTR + cc],
                   Ab + (size_t)row * lda + kk + cc);
    }
#pragma unroll
    for (int i = 0; i < 4; ++i) {  // B tile: 256 rows x 32 cols
      int c   = tid + i * 256;     // 0..1023
      int row = c >> 2;            // 0..255
      int cc  = (c & 3) << 3;
      async_copy16(&Bs[buf][row * LDSTR + cc],
                   Bb + (size_t)row * ldb + kk + cc);
    }
  };

  v8f acc[4][4];
#pragma unroll
  for (int mi = 0; mi < 4; ++mi)
#pragma unroll
    for (int ni = 0; ni < 4; ++ni)
#pragma unroll
      for (int j = 0; j < 8; ++j) acc[mi][ni][j] = 0.f;

  stage_tiles(0, 0);
  wait_async0();
  __syncthreads();

  for (int kk = 0; kk < K; kk += 32) {
    const int cur = (kk >> 5) & 1;
    if (kk + 32 < K) stage_tiles(cur ^ 1, kk + 32);  // DMA under the WMMAs
    v16bf af[4], bfv[4];
#pragma unroll
    for (int mi = 0; mi < 4; ++mi)
      af[mi] = frag_lds(&As[cur][(wr * 64 + mi * 16) * LDSTR], LDSTR, lane);
#pragma unroll
    for (int ni = 0; ni < 4; ++ni)
      bfv[ni] = frag_lds(&Bs[cur][(wc * 64 + ni * 16) * LDSTR], LDSTR, lane);
#pragma unroll
    for (int mi = 0; mi < 4; ++mi)
#pragma unroll
      for (int ni = 0; ni < 4; ++ni)
        acc[mi][ni] = wmma_bf16f32(af[mi], bfv[ni], acc[mi][ni]);
    wait_async0();
    __syncthreads();
  }

  // Epilogue: C layout per ISA: VGPR j -> row = j + 8*(lane>>4); col = lane&15
  const int rofs = (lane >> 4) << 3;
#pragma unroll
  for (int mi = 0; mi < 4; ++mi) {
#pragma unroll
    for (int ni = 0; ni < 4; ++ni) {
      size_t col = n0 + (size_t)wc * 64 + ni * 16 + (lane & 15);
#pragma unroll
      for (int j = 0; j < 8; ++j) {
        size_t row = m0 + (size_t)wr * 64 + mi * 16 + rofs + j;
        size_t idx = row * (size_t)ldc + col;
        float v = acc[mi][ni][j];
        if (Cbf) {
          Cbf[idx] = (__bf16)v;
        } else {
          if (accumulate) v += C[idx];
          C[idx] = v;
        }
      }
    }
  }
}

// ---------------------------------------------------------------------------
// RoPE + elu feature map: in f32 [row,512] -> out bf16 (elu(rope(x))+1)
// ---------------------------------------------------------------------------
__global__ __launch_bounds__(256) void rope_featmap_kernel(
    const float* __restrict__ in, __bf16* __restrict__ out) {
  const size_t row = blockIdx.x;
  const int i = threadIdx.x;              // pair index 0..255
  const int pos = (int)(row & (LL - 1));  // position within sequence
  const float ang =
      (float)pos * __expf((float)(2 * i) * (-9.2103403719761836f / 512.0f));
  float s, c;
  __sincosf(ang, &s, &c);
  const size_t base = row * DD + 2 * i;
  const float x0 = in[base], x1 = in[base + 1];
  const float r0 = x0 * c - x1 * s;
  const float r1 = x1 * c + x0 * s;
  out[base]     = (__bf16)(r0 > 0.f ? r0 + 1.f : __expf(r0));
  out[base + 1] = (__bf16)(r1 > 0.f ? r1 + 1.f : __expf(r1));
}

// ---------------------------------------------------------------------------
// Ksum[b,h,d] = sum_s Kf[b,s,h,d]     (one block per (b,h))
// ---------------------------------------------------------------------------
__global__ __launch_bounds__(256) void ksum_kernel(
    const __bf16* __restrict__ Kf, float* __restrict__ Ksum) {
  __shared__ float red[256];
  const int tid = threadIdx.x;
  const int bh = blockIdx.x, b = bh >> 3, h = bh & 7;
  const int d = tid & 63, grp = tid >> 6;
  float acc = 0.f;
  for (int s = grp; s < SSEQ; s += 4)
    acc += (float)Kf[((size_t)b * SSEQ + s) * DD + h * DHD + d];
  red[tid] = acc;
  __syncthreads();
  if (tid < 64)
    Ksum[(size_t)bh * DHD + d] =
        red[tid] + red[tid + 64] + red[tid + 128] + red[tid + 192];
}

// ---------------------------------------------------------------------------
// KVt[b,h,v,d] = sum_s Vb[b,s,h,v] * Kf[b,s,h,d]  (stored transposed, bf16)
// One block (8 waves) per (b,h); each wave owns 2 of the 16 16x16 tiles.
// ---------------------------------------------------------------------------
__global__ __launch_bounds__(256) void kv_kernel(
    const __bf16* __restrict__ Kf, const __bf16* __restrict__ Vb,
    __bf16* __restrict__ KVt) {
  __shared__ alignas(16) __bf16 Kt[64 * 72];  // [d][s_local]
  __shared__ alignas(16) __bf16 Vt[64 * 72];  // [v][s_local]
  const int tid = threadIdx.x, lane = tid & 31, wid = tid >> 5;
  const int bh = blockIdx.x, b = bh >> 3, h = bh & 7;
  const int t0 = wid * 2;

  v8f acc[2];
#pragma unroll
  for (int t = 0; t < 2; ++t)
#pragma unroll
    for (int j = 0; j < 8; ++j) acc[t][j] = 0.f;

  for (int s0 = 0; s0 < SSEQ; s0 += 64) {
    __syncthreads();
#pragma unroll
    for (int i = 0; i < 2; ++i) {
      int c = tid + i * 256;  // 0..511
      int srow = c >> 3;      // local s 0..63
      int d0 = (c & 7) << 3;  // 0..56
      size_t g = ((size_t)b * SSEQ + s0 + srow) * DD + h * DHD + d0;
      v8bf kk = *(const v8bf*)(Kf + g);
      v8bf vv = *(const v8bf*)(Vb + g);
#pragma unroll
      for (int e = 0; e < 8; ++e) {
        Kt[(d0 + e) * 72 + srow] = kk[e];
        Vt[(d0 + e) * 72 + srow] = vv[e];
      }
    }
    __syncthreads();
#pragma unroll
    for (int ks = 0; ks < 64; ks += 32) {
#pragma unroll
      for (int ti = 0; ti < 2; ++ti) {
        int t = t0 + ti;
        int mv = t >> 2, nd = t & 3;
        v16bf a  = frag_lds(&Vt[(mv * 16) * 72 + ks], 72, lane);
        v16bf bb = frag_lds(&Kt[(nd * 16) * 72 + ks], 72, lane);
        acc[ti] = wmma_bf16f32(a, bb, acc[ti]);
      }
    }
  }

  const int rofs = (lane >> 4) << 3;
#pragma unroll
  for (int ti = 0; ti < 2; ++ti) {
    int t = t0 + ti;
    int mv = t >> 2, nd = t & 3;
#pragma unroll
    for (int j = 0; j < 8; ++j) {
      int v = mv * 16 + rofs + j;
      int d = nd * 16 + (lane & 15);
      KVt[((size_t)bh * 64 + v) * 64 + d] = (__bf16)acc[ti][j];
    }
  }
}

// ---------------------------------------------------------------------------
// Z[b,l,h] = 1 / (where(score>thr, score, 0) + 1e-6), score = Q . Ksum
// (the reference's /s_len inside KV and *s_len on out cancel exactly)
// ---------------------------------------------------------------------------
__global__ __launch_bounds__(256) void scores_kernel(
    const __bf16* __restrict__ Qf, const float* __restrict__ Ksum,
    const float* __restrict__ thr, float* __restrict__ Zs) {
  size_t idx = (size_t)blockIdx.x * 256 + threadIdx.x;  // over B*L*H
  size_t bl = idx >> 3;
  int h = (int)(idx & 7);
  int b = (int)(bl / LL);
  const __bf16* q = Qf + bl * DD + h * DHD;
  const float* ks = Ksum + ((size_t)b * HH + h) * DHD;
  float sc = 0.f;
#pragma unroll 8
  for (int d = 0; d < DHD; ++d) sc += (float)q[d] * ks[d];
  float sp = sc > thr[0] ? sc : 0.f;
  Zs[idx] = 1.0f / (sp + 1e-6f);
}

// ---------------------------------------------------------------------------
// Ao[b,l,h*64+v] = Z[b,l,h] * sum_d Q[b,l,h,d] * KVt[b,h,v,d]   (bf16 out)
// Block: 8 waves; tile 128 rows x 64 cols; K=64 (2 wmma steps). Async staging.
// ---------------------------------------------------------------------------
__global__ __launch_bounds__(256) void attn_out_kernel(
    const __bf16* __restrict__ Qf, const __bf16* __restrict__ KVt,
    const float* __restrict__ Zs, __bf16* __restrict__ Ao) {
  __shared__ alignas(16) __bf16 Aq[128 * 72];
  __shared__ alignas(16) __bf16 Kv[64 * 72];
  __shared__ alignas(32) float zsh[128];
  const int tid = threadIdx.x, lane = tid & 31, wid = tid >> 5;
  const int bh = blockIdx.y, b = bh >> 3, h = bh & 7;
  const size_t rowbase = (size_t)b * LL + (size_t)blockIdx.x * 128;

#pragma unroll
  for (int i = 0; i < 4; ++i) {  // Q tile 128x64 (async DMA)
    int c = tid + i * 256;
    int row = c >> 3, cc = (c & 7) << 3;
    async_copy16(&Aq[row * 72 + cc], Qf + (rowbase + row) * DD + h * DHD + cc);
  }
#pragma unroll
  for (int i = 0; i < 2; ++i) {  // KVt 64x64 (async DMA)
    int c = tid + i * 256;
    int row = c >> 3, cc = (c & 7) << 3;
    async_copy16(&Kv[row * 72 + cc], KVt + ((size_t)bh * 64 + row) * 64 + cc);
  }
  if (tid < 128) zsh[tid] = Zs[(rowbase + tid) * HH + h];
  wait_async0();
  __syncthreads();

  v8f acc[4];
#pragma unroll
  for (int ni = 0; ni < 4; ++ni)
#pragma unroll
    for (int j = 0; j < 8; ++j) acc[ni][j] = 0.f;

#pragma unroll
  for (int ks = 0; ks < 64; ks += 32) {
    v16bf a = frag_lds(&Aq[(wid * 16) * 72 + ks], 72, lane);
#pragma unroll
    for (int ni = 0; ni < 4; ++ni) {
      v16bf bb = frag_lds(&Kv[(ni * 16) * 72 + ks], 72, lane);
      acc[ni] = wmma_bf16f32(a, bb, acc[ni]);
    }
  }

  const int rofs = (lane >> 4) << 3;
  v8f zv = *(const v8f*)(&zsh[wid * 16 + rofs]);
#pragma unroll
  for (int ni = 0; ni < 4; ++ni) {
    int col = h * DHD + ni * 16 + (lane & 15);
#pragma unroll
    for (int j = 0; j < 8; ++j) {
      size_t row = rowbase + wid * 16 + rofs + j;
      Ao[row * DD + col] = (__bf16)(acc[ni][j] * zv[j]);
    }
  }
}

// ---------------------------------------------------------------------------
// LayerNorm over 512 f32; writes f32 and/or bf16 outputs.
// ---------------------------------------------------------------------------
__global__ __launch_bounds__(256) void layernorm_kernel(
    const float* __restrict__ in, const float* __restrict__ g,
    const float* __restrict__ bta, float* __restrict__ out_f32,
    __bf16* __restrict__ out_bf16) {
  __shared__ float rs[256], rq[256];
  const size_t row = blockIdx.x;
  const int tid = threadIdx.x;
  const float* xr = in + row * DD;
  float v0 = xr[tid], v1 = xr[tid + 256];
  rs[tid] = v0 + v1;
  rq[tid] = v0 * v0 + v1 * v1;
  __syncthreads();
  for (int ofs = 128; ofs > 0; ofs >>= 1) {
    if (tid < ofs) {
      rs[tid] += rs[tid + ofs];
      rq[tid] += rq[tid + ofs];
    }
    __syncthreads();
  }
  const float mu = rs[0] * (1.0f / DD);
  const float var = rq[0] * (1.0f / DD) - mu * mu;
  const float rstd = rsqrtf(var + 1e-5f);
  float o0 = (v0 - mu) * rstd * g[tid] + bta[tid];
  float o1 = (v1 - mu) * rstd * g[tid + 256] + bta[tid + 256];
  if (out_f32) {
    out_f32[row * DD + tid] = o0;
    out_f32[row * DD + tid + 256] = o1;
  }
  if (out_bf16) {
    out_bf16[row * DD + tid] = (__bf16)o0;
    out_bf16[row * DD + tid + 256] = (__bf16)o1;
  }
}

// ---------------------------------------------------------------------------
// out = x + msg2 + LN(aug32 + baug; gaug, bln)
// ---------------------------------------------------------------------------
__global__ __launch_bounds__(256) void final_kernel(
    const float* __restrict__ x, const float* __restrict__ msg2,
    const float* __restrict__ aug32, const float* __restrict__ baug,
    const float* __restrict__ gaug, const float* __restrict__ bln,
    float* __restrict__ out) {
  __shared__ float rs[256], rq[256];
  const size_t row = blockIdx.x;
  const int tid = threadIdx.x;
  float a0 = aug32[row * DD + tid] + baug[tid];
  float a1 = aug32[row * DD + tid + 256] + baug[tid + 256];
  rs[tid] = a0 + a1;
  rq[tid] = a0 * a0 + a1 * a1;
  __syncthreads();
  for (int ofs = 128; ofs > 0; ofs >>= 1) {
    if (tid < ofs) {
      rs[tid] += rs[tid + ofs];
      rq[tid] += rq[tid + ofs];
    }
    __syncthreads();
  }
  const float mu = rs[0] * (1.0f / DD);
  const float var = rq[0] * (1.0f / DD) - mu * mu;
  const float rstd = rsqrtf(var + 1e-5f);
  float l0 = (a0 - mu) * rstd * gaug[tid] + bln[tid];
  float l1 = (a1 - mu) * rstd * gaug[tid + 256] + bln[tid + 256];
  out[row * DD + tid]       = x[row * DD + tid] + msg2[row * DD + tid] + l0;
  out[row * DD + tid + 256] =
      x[row * DD + tid + 256] + msg2[row * DD + tid + 256] + l1;
}

// ---------------------------------------------------------------------------
// Host orchestration
// ---------------------------------------------------------------------------
extern "C" void kernel_launch(void* const* d_in, const int* in_sizes, int n_in,
                              void* d_out, int out_size, void* d_ws,
                              size_t ws_size, hipStream_t stream) {
  (void)in_sizes; (void)n_in; (void)out_size; (void)ws_size;
  const float* x      = (const float*)d_in[0];
  const float* source = (const float*)d_in[1];
  const float* Wq     = (const float*)d_in[2];
  const float* Wk     = (const float*)d_in[3];
  const float* Wv     = (const float*)d_in[4];
  const float* Wmerge = (const float*)d_in[5];
  const float* Wmlp1  = (const float*)d_in[6];
  const float* Wmlp2  = (const float*)d_in[7];
  const float* g1     = (const float*)d_in[8];
  const float* b1     = (const float*)d_in[9];
  const float* g2     = (const float*)d_in[10];
  const float* b2     = (const float*)d_in[11];
  const float* Waug   = (const float*)d_in[12];
  const float* baug   = (const float*)d_in[13];
  const float* gaug   = (const float*)d_in[14];
  const float* bln    = (const float*)d_in[15];
  const float* thr    = (const float*)d_in[16];
  float* out = (float*)d_out;

  char* ws = (char*)d_ws;
  size_t off = 0;
  auto take = [&](size_t bytes) -> char* {
    char* p = ws + off;
    off += (bytes + 255) & ~(size_t)255;
    return p;
  };

  __bf16* xb   = (__bf16*)take((size_t)NT * DD * 2);
  __bf16* sb   = (__bf16*)take((size_t)NT * DD * 2);
  __bf16* WqT  = (__bf16*)take((size_t)DD * DD * 2);
  __bf16* WkT  = (__bf16*)take((size_t)DD * DD * 2);
  __bf16* WvT  = (__bf16*)take((size_t)DD * DD * 2);
  __bf16* WmT  = (__bf16*)take((size_t)DD * DD * 2);
  __bf16* W1T  = (__bf16*)take((size_t)D2 * D2 * 2);
  __bf16* W2T  = (__bf16*)take((size_t)DD * D2 * 2);
  __bf16* WaT  = (__bf16*)take((size_t)DD * DD * 2);
  float*  C1   = (float*)take((size_t)NT * D2 * 4);
  __bf16* Qf   = (__bf16*)take((size_t)NT * DD * 2);
  __bf16* Kf   = (__bf16*)take((size_t)NT * DD * 2);
  __bf16* Vb   = (__bf16*)take((size_t)NT * DD * 2);
  float*  Ksum = (float*)take((size_t)BB * HH * DHD * 4);
  __bf16* KVt  = (__bf16*)take((size_t)BB * HH * 64 * 64 * 2);
  float*  Zs   = (float*)take((size_t)NT * HH * 4);
  __bf16* Ao   = (__bf16*)take((size_t)NT * DD * 2);
  __bf16* Msgb = (__bf16*)take((size_t)NT * DD * 2);
  __bf16* Hb   = (__bf16*)take((size_t)NT * D2 * 2);
  float*  Msg2 = (float*)take((size_t)NT * DD * 4);

  const dim3 blk(256);
  const size_t nXD = (size_t)NT * DD;   // 8.39M
  const size_t nXD2 = (size_t)NT * D2;  // 16.78M
  auto blocks = [](size_t n) { return dim3((unsigned)((n + 255) / 256)); };

  // 1) Convert activations and (transposed) weights to bf16
  f32_to_bf16_kernel<<<blocks(nXD), blk, 0, stream>>>(x, xb, nXD, 0);
  f32_to_bf16_kernel<<<blocks(nXD), blk, 0, stream>>>(source, sb, nXD, 0);
  transpose_bf16_kernel<<<blocks((size_t)DD * DD), blk, 0, stream>>>(Wq, WqT, DD, DD);
  transpose_bf16_kernel<<<blocks((size_t)DD * DD), blk, 0, stream>>>(Wk, WkT, DD, DD);
  transpose_bf16_kernel<<<blocks((size_t)DD * DD), blk, 0, stream>>>(Wv, WvT, DD, DD);
  transpose_bf16_kernel<<<blocks((size_t)DD * DD), blk, 0, stream>>>(Wmerge, WmT, DD, DD);
  transpose_bf16_kernel<<<blocks((size_t)D2 * D2), blk, 0, stream>>>(Wmlp1, W1T, D2, D2);
  transpose_bf16_kernel<<<blocks((size_t)D2 * DD), blk, 0, stream>>>(Wmlp2, W2T, D2, DD);
  transpose_bf16_kernel<<<blocks((size_t)DD * DD), blk, 0, stream>>>(Waug, WaT, DD, DD);

  // 2) mq = x@Wq -> RoPE+elu -> Qf
  gemm_bf16_kernel<<<dim3(DD / 256, NT / 128), blk, 0, stream>>>(
      xb, DD, WqT, DD, C1, DD, DD, 0, nullptr);
  rope_featmap_kernel<<<dim3(NT), blk, 0, stream>>>(C1, Qf);

  // 3) mk = source@Wk -> RoPE+elu -> Kf
  gemm_bf16_kernel<<<dim3(DD / 256, NT / 128), blk, 0, stream>>>(
      sb, DD, WkT, DD, C1, DD, DD, 0, nullptr);
  rope_featmap_kernel<<<dim3(NT), blk, 0, stream>>>(C1, Kf);

  // 4) v = source@Wv -> Vb (bf16, fused epilogue)
  gemm_bf16_kernel<<<dim3(DD / 256, NT / 128), blk, 0, stream>>>(
      sb, DD, WvT, DD, C1, DD, DD, 0, Vb);

  // 5) attention core
  ksum_kernel<<<dim3(BB * HH), blk, 0, stream>>>(Kf, Ksum);
  kv_kernel<<<dim3(BB * HH), blk, 0, stream>>>(Kf, Vb, KVt);
  scores_kernel<<<dim3((NT * HH) / 256), blk, 0, stream>>>(Qf, Ksum, thr, Zs);
  attn_out_kernel<<<dim3(LL / 128, BB * HH), blk, 0, stream>>>(Qf, KVt, Zs, Ao);

  // 6) merge + LN1 -> Msgb (bf16)
  gemm_bf16_kernel<<<dim3(DD / 256, NT / 128), blk, 0, stream>>>(
      Ao, DD, WmT, DD, C1, DD, DD, 0, nullptr);
  layernorm_kernel<<<dim3(NT), blk, 0, stream>>>(C1, g1, b1, nullptr, Msgb);

  // 7) MLP1: h = [x, msg] @ Wmlp1 = x@W1[:512] + msg@W1[512:]
  gemm_bf16_kernel<<<dim3(D2 / 256, NT / 128), blk, 0, stream>>>(
      xb, DD, W1T, D2, C1, D2, DD, 0, nullptr);
  gemm_bf16_kernel<<<dim3(D2 / 256, NT / 128), blk, 0, stream>>>(
      Msgb, DD, W1T + DD, D2, C1, D2, DD, 1, nullptr);
  f32_to_bf16_kernel<<<blocks(nXD2), blk, 0, stream>>>(C1, Hb, nXD2, 1);  // relu

  // 8) MLP2 + LN2 -> Msg2 (f32)
  gemm_bf16_kernel<<<dim3(DD / 256, NT / 128), blk, 0, stream>>>(
      Hb, D2, W2T, D2, C1, DD, D2, 0, nullptr);
  layernorm_kernel<<<dim3(NT), blk, 0, stream>>>(C1, g2, b2, Msg2, nullptr);

  // 9) aug GEMM, then fused LN(aug)+residual
  gemm_bf16_kernel<<<dim3(DD / 256, NT / 128), blk, 0, stream>>>(
      xb, DD, WaT, DD, C1, DD, DD, 0, nullptr);
  final_kernel<<<dim3(NT), blk, 0, stream>>>(x, Msg2, C1, baug, gaug, bln, out);
}
